// Streaming_61572651156213
// MI455X (gfx1250) — compile-verified, matrix operand order checked by reading, not run on hardware
//
#include <hip/hip_runtime.h>
#include <stdint.h>

// ---------------------------------------------------------------------------
// Streaming top-k retrieval for MI455X (gfx1250, wave32).
//   queries   : [512][128]  f32
//   candidates: [1048576][128] f32
//   identifiers:[1048576] i32, k = 100
// d_out: [512*100] f32 scores, then [512*100] i32 ids (bitcast).
//
// GEMM precision: split-bf16 (hi = truncate-to-bf16, lo = x - hi).
//   Q.C = Ah.Bh + Ah.Bl + Al.Bh  (+ O(2^-16) dropped Al.Bl term)
// -> 3x v_wmma_f32_16x16x32_bf16 per K=32 vs 8x v_wmma_f32_16x16x4_f32.
// ---------------------------------------------------------------------------

typedef float v8f __attribute__((ext_vector_type(8)));
typedef __bf16 v16bf __attribute__((ext_vector_type(16)));
typedef unsigned int u32x4 __attribute__((ext_vector_type(4)));
typedef int i32x4 __attribute__((ext_vector_type(4)));
typedef int i32x8 __attribute__((ext_vector_type(8)));

#define DIM   128     // feature dim
#define NQ    512     // queries
#define KSEL  100     // top-k
#define CT    64      // candidates per LDS tile
#define SCC   16384   // candidates per superchunk
#define NSC   64      // superchunks
#define QT    128     // queries per block (8 waves * 16)

__device__ __forceinline__ unsigned long long pack_key(float s, unsigned int gc) {
    unsigned int u = __float_as_uint(s);
    u = (u & 0x80000000u) ? ~u : (u | 0x80000000u);
    return ((unsigned long long)u << 32) | (unsigned long long)(~gc);
}

// Split 4 f32 into packed bf16 hi/lo pairs (truncation split; integer-only).
__device__ __forceinline__ void cvt4(float4 x, unsigned* hw, unsigned* lw) {
    unsigned b1 = __float_as_uint(x.y), b3 = __float_as_uint(x.w);
    unsigned h0 = __float_as_uint(x.x) & 0xFFFF0000u;
    unsigned h1 = b1 & 0xFFFF0000u;
    unsigned h2 = __float_as_uint(x.z) & 0xFFFF0000u;
    unsigned h3 = b3 & 0xFFFF0000u;
    unsigned l0 = __float_as_uint(x.x - __uint_as_float(h0));
    unsigned l1 = __float_as_uint(x.y - __uint_as_float(h1));
    unsigned l2 = __float_as_uint(x.z - __uint_as_float(h2));
    unsigned l3 = __float_as_uint(x.w - __uint_as_float(h3));
    hw[0] = (h0 >> 16) | h1;
    hw[1] = (h2 >> 16) | h3;
    lw[0] = (l0 >> 16) | (l1 & 0xFFFF0000u);
    lw[1] = (l2 >> 16) | (l3 & 0xFFFF0000u);
}

__device__ __forceinline__ v16bf loadfrag(const unsigned* base) {
    union { uint4 q[2]; v16bf v; } u;
    u.q[0] = *(const uint4*)(base);
    u.q[1] = *(const uint4*)(base + 8);
    return u.v;
}

// ---------------------------------------------------------------------------
// Phase 1: TDM-staged split-bf16 WMMA GEMM + per-superchunk top-100/query.
// grid = (4, 64); block = 256 threads (8 waves).
// ---------------------------------------------------------------------------
extern "C" __global__ void __launch_bounds__(256, 1)
topk_phase1(const float* __restrict__ Q,
            const float* __restrict__ Cand,
            unsigned long long* __restrict__ pool /* [NSC][NQ][KSEL] */)
{
    __shared__ float stage[2][CT * DIM];                 // 2 x 32 KB (TDM dst)
    __shared__ unsigned int Bh[CT * (DIM / 2)];          // 16 KB packed bf16 hi
    __shared__ unsigned int Bl[CT * (DIM / 2)];          // 16 KB packed bf16 lo
    __shared__ unsigned long long lists[8 * 32 * 8 * 8]; // 128 KB per-lane lists

    const int tid  = threadIdx.x;
    const int wave = tid >> 5;
    const int lane = tid & 31;
    const int m    = lane & 15;       // row within 16 (A) / col within 16 (B)
    const int h    = lane >> 4;       // half-wave select
    const int q0   = blockIdx.x * QT + wave * 16;
    const long long cbase = (long long)blockIdx.y * SCC;

    // per-lane top-8 lists (8 C-rows each), plus register threshold cache
    unsigned long long* lbase = &lists[(size_t)(wave * 32 + lane) * 64];
    unsigned long long thr[8];
    #pragma unroll
    for (int i = 0; i < 64; ++i) lbase[i] = 0ull;
    #pragma unroll
    for (int r = 0; r < 8; ++r) thr[r] = 0ull;

    // ---- A fragments: split-bf16, all K resident (8 x v16bf = 64 VGPRs) ---
    // bf16 A layout: lane half h, VGPR v -> K = 32s + 8h + (v<4 ? 2v : 16+2(v-4))
    v16bf a_hi[4], a_lo[4];
    {
        const float4* Q4 = (const float4*)&Q[(size_t)(q0 + m) * DIM];
        #pragma unroll
        for (int s = 0; s < 4; ++s) {
            unsigned hw[8], lw[8];
            cvt4(Q4[8 * s + 2 * h],     &hw[0], &lw[0]);
            cvt4(Q4[8 * s + 2 * h + 1], &hw[2], &lw[2]);
            cvt4(Q4[8 * s + 4 + 2 * h],     &hw[4], &lw[4]);
            cvt4(Q4[8 * s + 4 + 2 * h + 1], &hw[6], &lw[6]);
            union { unsigned w[8]; v16bf v; } uh, ul;
            #pragma unroll
            for (int i = 0; i < 8; ++i) { uh.w[i] = hw[i]; ul.w[i] = lw[i]; }
            a_hi[s] = uh.v;
            a_lo[s] = ul.v;
        }
    }

    // -------- Tensor Data Mover: stage 64x128 f32 tile into LDS ------------
    const unsigned int lds_base = (unsigned int)(uintptr_t)(&stage[0][0]);
    auto issue_tdm = [&](int buf, int t) {
        unsigned long long ga =
            (unsigned long long)(uintptr_t)(Cand + (cbase + (long long)t * CT) * DIM);
        u32x4 g0;
        g0[0] = 1u;                                    // count=1
        g0[1] = lds_base + (unsigned int)buf * (CT * DIM * 4);
        g0[2] = (unsigned int)ga;
        g0[3] = (unsigned int)(ga >> 32) | (2u << 30); // type=2
        i32x8 g1;
        g1[0] = (int)(2u << 16);       // data_size = 4B
        g1[1] = (int)(128u << 16);     // tensor_dim0 = 128
        g1[2] = (int)(16384u << 16);   // tensor_dim1 = 16384
        g1[3] = (int)(128u << 16);     // tile_dim0 = 128
        g1[4] = (int)64u;              // tile_dim1 = 64
        g1[5] = (int)128u;             // tensor_dim0_stride = 128
        g1[6] = 0; g1[7] = 0;
        i32x4 gz4 = {0, 0, 0, 0};
        i32x8 gz8 = {0, 0, 0, 0, 0, 0, 0, 0};
        __builtin_amdgcn_tensor_load_to_lds(g0, g1, gz4, gz4, gz8, 0);
    };

    if (wave == 0) {
        issue_tdm(0, 0);
        __builtin_amdgcn_s_wait_tensorcnt(0);
    }
    __syncthreads();

    // per-tile selection update into this lane's lists
    auto update = [&](const v8f& c, unsigned int gc) {
        #pragma unroll
        for (int r = 0; r < 8; ++r) {
            unsigned long long key = pack_key(c[r], gc);
            if (key > thr[r]) {
                unsigned long long* Lr = lbase + r * 8;
                unsigned long long v[8];
                #pragma unroll
                for (int e = 0; e < 8; ++e) v[e] = Lr[e];
                #pragma unroll
                for (int e = 7; e >= 1; --e)
                    v[e] = (key > v[e]) ? ((key > v[e - 1]) ? v[e - 1] : key) : v[e];
                v[0] = (key > v[0]) ? key : v[0];
                #pragma unroll
                for (int e = 0; e < 8; ++e) Lr[e] = v[e];
                thr[r] = v[7];
            }
        }
    };

    const int NT = SCC / CT;  // 256 tiles
    int cur = 0;
    for (int t = 0; t < NT; ++t) {
        if (wave == 0 && t + 1 < NT) issue_tdm(cur ^ 1, t + 1);

        // ---- cooperative f32 -> packed split-bf16 conversion of the tile --
        {
            const int row = tid >> 2;           // 0..63
            const int ks  = (tid & 3) * 32;     // k-segment base
            const float4* S4 = (const float4*)&stage[cur][row * DIM + ks];
            unsigned hw[16], lw[16];
            #pragma unroll
            for (int g = 0; g < 8; ++g) cvt4(S4[g], &hw[g * 2], &lw[g * 2]);
            unsigned* dh = &Bh[row * 64 + ks / 2];
            unsigned* dl = &Bl[row * 64 + ks / 2];
            #pragma unroll
            for (int g = 0; g < 4; ++g) {
                uint4 th = { hw[g*4+0], hw[g*4+1], hw[g*4+2], hw[g*4+3] };
                uint4 tl = { lw[g*4+0], lw[g*4+1], lw[g*4+2], lw[g*4+3] };
                *(uint4*)(dh + g * 4) = th;
                *(uint4*)(dl + g * 4) = tl;
            }
        }
        __syncthreads();  // bf16 tile ready

        // ---- compute: 16 queries x 64 candidates, two accumulator chains --
        #pragma unroll
        for (int jp = 0; jp < 2; ++jp) {
            v8f c0 = {}, c1 = {};
            const int n0 = (jp * 2) * 16 + m;
            const int n1 = (jp * 2 + 1) * 16 + m;
            #pragma unroll
            for (int s = 0; s < 4; ++s) {
                const int off = s * 16 + 4 * h;
                v16bf bh0 = loadfrag(&Bh[n0 * 64 + off]);
                v16bf bl0 = loadfrag(&Bl[n0 * 64 + off]);
                v16bf bh1 = loadfrag(&Bh[n1 * 64 + off]);
                v16bf bl1 = loadfrag(&Bl[n1 * 64 + off]);
                c0 = __builtin_amdgcn_wmma_f32_16x16x32_bf16(
                         false, a_hi[s], false, bh0, (short)0, c0, false, false);
                c1 = __builtin_amdgcn_wmma_f32_16x16x32_bf16(
                         false, a_hi[s], false, bh1, (short)0, c1, false, false);
                c0 = __builtin_amdgcn_wmma_f32_16x16x32_bf16(
                         false, a_hi[s], false, bl0, (short)0, c0, false, false);
                c1 = __builtin_amdgcn_wmma_f32_16x16x32_bf16(
                         false, a_hi[s], false, bl1, (short)0, c1, false, false);
                c0 = __builtin_amdgcn_wmma_f32_16x16x32_bf16(
                         false, a_lo[s], false, bh0, (short)0, c0, false, false);
                c1 = __builtin_amdgcn_wmma_f32_16x16x32_bf16(
                         false, a_lo[s], false, bh1, (short)0, c1, false, false);
            }
            const unsigned int gcb =
                (unsigned int)(cbase + (long long)t * CT) + m;
            update(c0, gcb + (jp * 2) * 16);
            update(c1, gcb + (jp * 2 + 1) * 16);
        }

        if (wave == 0 && t + 1 < NT) __builtin_amdgcn_s_wait_tensorcnt(0);
        __syncthreads();  // compute done; next stage buffer landed
        cur ^= 1;
    }

    // ---- per-query top-100 of the 128-entry pooled lists (wave-local) -----
    for (int qi = 0; qi < 16; ++qi) {
        const int hh = qi >> 3;
        const int rr = qi & 7;
        unsigned long long v0, v1, v2, v3;
        {
            int f;
            f = lane;      v0 = lists[((size_t)(wave * 32 + hh * 16 + (f >> 3)) * 8 + rr) * 8 + (f & 7)];
            f = lane + 32; v1 = lists[((size_t)(wave * 32 + hh * 16 + (f >> 3)) * 8 + rr) * 8 + (f & 7)];
            f = lane + 64; v2 = lists[((size_t)(wave * 32 + hh * 16 + (f >> 3)) * 8 + rr) * 8 + (f & 7)];
            f = lane + 96; v3 = lists[((size_t)(wave * 32 + hh * 16 + (f >> 3)) * 8 + rr) * 8 + (f & 7)];
        }
        unsigned long long* out =
            &pool[((size_t)blockIdx.y * NQ + (q0 + qi)) * KSEL];
        for (int r = 0; r < KSEL; ++r) {
            unsigned long long mx = v0 > v1 ? v0 : v1;
            unsigned long long m2 = v2 > v3 ? v2 : v3;
            mx = mx > m2 ? mx : m2;
            #pragma unroll
            for (int off = 16; off >= 1; off >>= 1) {
                unsigned long long o = __shfl_xor(mx, off, 32);
                mx = o > mx ? o : mx;
            }
            if (lane == 0) out[r] = mx;
            if      (v0 == mx) v0 = 0;
            else if (v1 == mx) v1 = 0;
            else if (v2 == mx) v2 = 0;
            else if (v3 == mx) v3 = 0;
        }
    }
}

// ---------------------------------------------------------------------------
// Phase 2: merge NSC*KSEL = 6400 pooled keys per query -> final top-100.
// ---------------------------------------------------------------------------
extern "C" __global__ void __launch_bounds__(256, 1)
topk_phase2(const unsigned long long* __restrict__ pool,
            const int* __restrict__ ident,
            float* __restrict__ out_scores,
            int* __restrict__ out_ids)
{
    __shared__ unsigned long long sp[NSC * KSEL]; // 51.2 KB
    __shared__ unsigned long long best;
    const int q = blockIdx.x;
    const int tid = threadIdx.x;

    for (int i = tid; i < NSC * KSEL; i += 256) {
        int sc = i / KSEL, e = i - sc * KSEL;
        sp[i] = pool[((size_t)sc * NQ + q) * KSEL + e];
    }
    __syncthreads();

    for (int r = 0; r < KSEL; ++r) {
        if (tid == 0) best = 0ull;
        __syncthreads();
        unsigned long long lm = 0ull;
        for (int i = tid; i < NSC * KSEL; i += 256)
            lm = sp[i] > lm ? sp[i] : lm;
        if (lm) atomicMax(&best, lm);   // ds_max_rtn_u64 on LDS
        __syncthreads();
        const unsigned long long w = best;
        for (int i = tid; i < NSC * KSEL; i += 256)
            if (sp[i] == w) sp[i] = 0ull;
        if (tid == 0) {
            unsigned int u = (unsigned int)(w >> 32);
            unsigned int bits = (u & 0x80000000u) ? (u ^ 0x80000000u) : ~u;
            unsigned int gc = ~(unsigned int)(w & 0xFFFFFFFFu);
            out_scores[q * KSEL + r] = __uint_as_float(bits);
            out_ids[q * KSEL + r] = ident[gc];
        }
        __syncthreads();
    }
}

// ---------------------------------------------------------------------------
extern "C" void kernel_launch(void* const* d_in, const int* in_sizes, int n_in,
                              void* d_out, int out_size, void* d_ws, size_t ws_size,
                              hipStream_t stream) {
    const float* Q     = (const float*)d_in[0];
    const float* Cand  = (const float*)d_in[1];
    const int*   Ident = (const int*)d_in[2];
    // d_in[3] = k (constant 100)

    unsigned long long* pool = (unsigned long long*)d_ws; // 26.2 MB
    float* out_scores = (float*)d_out;
    int*   out_ids    = (int*)d_out + NQ * KSEL;

    dim3 g1(NQ / QT, NSC), b1(256);
    topk_phase1<<<g1, b1, 0, stream>>>(Q, Cand, pool);
    topk_phase2<<<NQ, 256, 0, stream>>>(pool, Ident, out_scores, out_ids);
}
